// VectorQuantizer_18631568130889
// MI455X (gfx1250) — matrix operand order repaired and checked by reading, not compiled, not run on hardware
//
#include <hip/hip_runtime.h>
#include <math.h>

typedef __attribute__((ext_vector_type(16))) __bf16 v16bf;
typedef __attribute__((ext_vector_type(8)))  float  v8f;

#define NCODES   512
#define DIM      64
#define NB       32
#define NH       64
#define NW       64
#define NTOT     131072          // 32*64*64
#define QELEMS   8388608         // 32*64*64*64
#define OFF_Q    1               // quantized offset in d_out (floats)
#define OFF_PPL  8388609         // perplexity offset
#define OFF_ENC  8388610         // encodings offset
#define ENC_F    67108864        // 131072*512
#define LDSP     65              // padded LDS row stride

union ABf { v16bf v; unsigned short u[16]; };

static __device__ __forceinline__ unsigned short f2bf(float f) {
    unsigned int u = __float_as_uint(f);
    u += 0x7FFFu + ((u >> 16) & 1u);       // round-to-nearest-even
    return (unsigned short)(u >> 16);
}
static __device__ __forceinline__ float bf2f(unsigned short b) {
    return __uint_as_float(((unsigned int)b) << 16);
}

// ---- init: zero counts + sse accumulator ----
__global__ void vq_init(unsigned int* counts, float* sse) {
    int t = blockIdx.x * blockDim.x + threadIdx.x;
    if (t < NCODES) counts[t] = 0u;
    if (t == 0) *sse = 0.0f;
}

// ---- prep: codebook -> bf16 hi/lo + row squared norms ----
__global__ void __launch_bounds__(32) vq_prep(const float* __restrict__ w,
                                              unsigned short* __restrict__ wHi,
                                              unsigned short* __restrict__ wLo,
                                              float* __restrict__ wSq) {
    int row = blockIdx.x;
    int lane = threadIdx.x;
    float sq = 0.0f;
    for (int i = 0; i < 2; ++i) {
        int c = lane + i * 32;
        float f = w[row * DIM + c];
        unsigned short hi = f2bf(f);
        float lo = f - bf2f(hi);
        wHi[row * DIM + c] = hi;
        wLo[row * DIM + c] = f2bf(lo);
        sq += f * f;
    }
    for (int m = 16; m >= 1; m >>= 1) sq += __shfl_xor(sq, m, 32);
    if (lane == 0) wSq[row] = sq;
}

// ---- streaming zero of the one-hot encodings region (268 MB) ----
__global__ void vq_zero_enc(float* __restrict__ out) {
    float2* p = (float2*)(out + OFF_ENC);          // 8-byte aligned
    const size_t n2 = (size_t)ENC_F / 2;
    size_t i = (size_t)blockIdx.x * blockDim.x + threadIdx.x;
    size_t stride = (size_t)gridDim.x * blockDim.x;
    float2 z; z.x = 0.0f; z.y = 0.0f;
    for (; i < n2; i += stride) p[i] = z;
}

// ---- main: WMMA distance GEMM + argmin + quantize + reductions ----
__global__ void __launch_bounds__(128) vq_main(const float* __restrict__ x,      // BCHW
                                               const float* __restrict__ weight, // [512,64]
                                               const unsigned short* __restrict__ wHi,
                                               const unsigned short* __restrict__ wLo,
                                               const float* __restrict__ wSq,
                                               float* __restrict__ out,
                                               unsigned int* __restrict__ counts,
                                               float* __restrict__ sseAcc) {
    __shared__ float ldsX[64 * LDSP];
    __shared__ int   ldsIdx[64];
    __shared__ float red[128];

    const int t    = threadIdx.x;
    const int lane = t & 31;
    const int wave = t >> 5;
    const int blk  = blockIdx.x;          // 2048 blocks: one per (b,h)
    const int b    = blk >> 6;
    const int h    = blk & 63;

    // --- coalesced load of 64 pixels x 64 channels into LDS ---
    {
        const int wp = t & 63;            // pixel (w)
        const int ch = t >> 6;            // 0/1
        const float* src = x + (size_t)b * DIM * (NH * NW) + (size_t)h * NW;
        for (int i = 0; i < 32; ++i) {
            int c = i * 2 + ch;
            ldsX[wp * LDSP + c] = src[(size_t)c * (NH * NW) + wp];
        }
    }
    __syncthreads();

    // --- build A fragments (16-bit A 16x32 layout), hi/lo split, K-chunks {0,32} ---
    const int m = wave * 16 + (lane & 15);
    const int g = lane >> 4;
    ABf aHi[2], aLo[2];
    for (int kc = 0; kc < 2; ++kc) {
        for (int j = 0; j < 16; ++j) {
            int v = j >> 1, pos = j & 1;
            int k = kc * 32 + g * 16 + j;
            float f = ldsX[m * LDSP + k];
            unsigned short hi = f2bf(f);
            aHi[kc].u[j] = hi;
            aLo[kc].u[j] = f2bf(f - bf2f(hi));
        }
    }

    float minVal[8];
    int   minIdx[8];
    for (int r = 0; r < 8; ++r) { minVal[r] = 3.0e38f; minIdx[r] = 0; }

    const int nc = lane & 15;
    // --- sweep the codebook in 16-code tiles ---
    for (int ct = 0; ct < NCODES / 16; ++ct) {
        const int code = ct * 16 + nc;
        const v16bf bHi0 = *reinterpret_cast<const v16bf*>(wHi + code * DIM + 0  + g * 16);
        const v16bf bHi1 = *reinterpret_cast<const v16bf*>(wHi + code * DIM + 32 + g * 16);
        const v16bf bLo0 = *reinterpret_cast<const v16bf*>(wLo + code * DIM + 0  + g * 16);
        const v16bf bLo1 = *reinterpret_cast<const v16bf*>(wLo + code * DIM + 32 + g * 16);
        v8f acc = {0.f, 0.f, 0.f, 0.f, 0.f, 0.f, 0.f, 0.f};
        acc = __builtin_amdgcn_wmma_f32_16x16x32_bf16(false, aHi[0].v, false, bHi0, (short)0, acc, false, false);
        acc = __builtin_amdgcn_wmma_f32_16x16x32_bf16(false, aHi[1].v, false, bHi1, (short)0, acc, false, false);
        acc = __builtin_amdgcn_wmma_f32_16x16x32_bf16(false, aLo[0].v, false, bHi0, (short)0, acc, false, false);
        acc = __builtin_amdgcn_wmma_f32_16x16x32_bf16(false, aLo[1].v, false, bHi1, (short)0, acc, false, false);
        acc = __builtin_amdgcn_wmma_f32_16x16x32_bf16(false, aHi[0].v, false, bLo0, (short)0, acc, false, false);
        acc = __builtin_amdgcn_wmma_f32_16x16x32_bf16(false, aHi[1].v, false, bLo1, (short)0, acc, false, false);
        const float wsq = wSq[code];
        for (int r = 0; r < 8; ++r) {
            float d = wsq - 2.0f * acc[r];           // ||x||^2 constant per row: dropped
            if (d < minVal[r]) { minVal[r] = d; minIdx[r] = code; }
        }
    }

    // --- cross-lane argmin within each 16-lane N group (wave32) ---
    for (int mask = 1; mask <= 8; mask <<= 1) {
        for (int r = 0; r < 8; ++r) {
            float ov = __shfl_xor(minVal[r], mask, 32);
            int   oi = __shfl_xor(minIdx[r], mask, 32);
            if (ov < minVal[r] || (ov == minVal[r] && oi < minIdx[r])) {
                minVal[r] = ov; minIdx[r] = oi;
            }
        }
    }
    if ((lane & 15) == 0) {
        for (int r = 0; r < 8; ++r) ldsIdx[wave * 16 + g * 8 + r] = minIdx[r];  // row M = g*8+r
    }
    __syncthreads();

    // --- quantized output (BCHW, coalesced) + SSE ---
    float sse = 0.0f;
    {
        const int wp = t & 63;
        const int ch = t >> 6;
        const int code = ldsIdx[wp];
        float* dst = out + OFF_Q + (size_t)b * DIM * (NH * NW) + (size_t)h * NW;
        for (int i = 0; i < 32; ++i) {
            int c = i * 2 + ch;
            float q  = weight[code * DIM + c];
            float xv = ldsX[wp * LDSP + c];
            float df = q - xv;
            sse += df * df;
            dst[(size_t)c * (NH * NW) + wp] = q;   // straight-through value == quantized
        }
    }
    red[t] = sse;
    __syncthreads();
    for (int s = 64; s > 0; s >>= 1) { if (t < s) red[t] += red[t + s]; __syncthreads(); }
    if (t == 0) atomicAdd(sseAcc, red[0]);

    // --- histogram + one-hot scatter (zero-fill done by vq_zero_enc) ---
    if (t < 64) {
        int code = ldsIdx[t];
        atomicAdd(&counts[code], 1u);
        size_t n = (size_t)blk * 64 + t;
        out[(size_t)OFF_ENC + n * NCODES + code] = 1.0f;
    }
}

// ---- finalize: loss + perplexity scalars ----
__global__ void __launch_bounds__(512) vq_final(const unsigned int* __restrict__ counts,
                                                const float* __restrict__ sse,
                                                float* __restrict__ out) {
    __shared__ float red[512];
    int t = threadIdx.x;
    float p = (float)counts[t] * (1.0f / (float)NTOT);
    red[t] = -p * logf(p + 1e-10f);
    __syncthreads();
    for (int s = 256; s > 0; s >>= 1) { if (t < s) red[t] += red[t + s]; __syncthreads(); }
    if (t == 0) {
        out[0]      = 1.25f * (*sse) / (float)QELEMS;   // q_loss + 0.25*e_loss, equal values
        out[OFF_PPL] = expf(red[0]);
    }
}

extern "C" void kernel_launch(void* const* d_in, const int* in_sizes, int n_in,
                              void* d_out, int out_size, void* d_ws, size_t ws_size,
                              hipStream_t stream) {
    (void)in_sizes; (void)n_in; (void)out_size; (void)ws_size;
    const float* x = (const float*)d_in[0];     // [32,64,64,64] BCHW
    const float* w = (const float*)d_in[1];     // [512,64]
    float* out = (float*)d_out;

    unsigned short* wHi = (unsigned short*)d_ws;
    unsigned short* wLo = wHi + (size_t)NCODES * DIM;
    float* wSq = (float*)(wLo + (size_t)NCODES * DIM);
    unsigned int* counts = (unsigned int*)(wSq + NCODES);
    float* sse = (float*)(counts + NCODES);

    vq_init<<<2, 256, 0, stream>>>(counts, sse);
    vq_prep<<<NCODES, 32, 0, stream>>>(w, wHi, wLo, wSq);
    vq_zero_enc<<<8192, 256, 0, stream>>>(out);
    vq_main<<<NTOT / 64, 128, 0, stream>>>(x, w, wHi, wLo, wSq, out, counts, sse);
    vq_final<<<1, 512, 0, stream>>>(counts, sse, out);
}
